// GNN_2705829396900
// MI455X (gfx1250) — matrix operand order repaired and checked by reading, not compiled, not run on hardware
//
#include <hip/hip_runtime.h>
#include <hip/hip_bf16.h>

#define NNODES 8192

typedef __attribute__((ext_vector_type(16))) __bf16 bf16x16;
typedef __attribute__((ext_vector_type(8)))  __bf16 bf16x8;
typedef __attribute__((ext_vector_type(8)))  float  v8f;

// ---------------------------------------------------------------------------
// K0a: extract eta/phi columns from x (row stride 7) into contiguous arrays
// ---------------------------------------------------------------------------
__global__ __launch_bounds__(256) void extract_kernel(
    const float* __restrict__ x, float* __restrict__ eta, float* __restrict__ phi)
{
    int i = blockIdx.x * 256 + threadIdx.x;
    if (i < NNODES) {
        eta[i] = x[i * 7 + 1];
        phi[i] = x[i * 7 + 2];
    }
}

// ---------------------------------------------------------------------------
// K0b: edge matrix. Row max of exp(-a*dR^2) is exactly 1 (diagonal), so the
// softmax numerator is exp(a_ij - 1). Store numerators bf16 (128MB, fits L2),
// rowsum in f32 (accumulated over the bf16-rounded values for consistency).
// ---------------------------------------------------------------------------
__global__ __launch_bounds__(256) void edges_kernel(
    const float* __restrict__ eta, const float* __restrict__ phi,
    const float* __restrict__ alpha, __bf16* __restrict__ Wm,
    float* __restrict__ R)
{
    const int i = blockIdx.x;
    const float ei = eta[i], pii = phi[i], al = alpha[0];
    const float PI_F = 3.14159265358979323846f;
    float partial = 0.0f;
    __bf16* wrow = Wm + (size_t)i * NNODES;
    for (int j = threadIdx.x; j < NNODES; j += 256) {
        float de = ei - eta[j];
        float dp = fabsf(pii - phi[j]);
        dp = (dp > PI_F) ? (2.0f * PI_F - dp) : dp;
        float d2 = de * de + dp * dp;
        float a  = __expf(-al * d2);
        float nm = __expf(a - 1.0f);
        __bf16 nb = (__bf16)nm;
        wrow[j] = nb;
        partial += (float)nb;
    }
    __shared__ float red[256];
    red[threadIdx.x] = partial;
    __syncthreads();
    for (int s = 128; s > 0; s >>= 1) {
        if (threadIdx.x < s) red[threadIdx.x] += red[threadIdx.x + s];
        __syncthreads();
    }
    if (threadIdx.x == 0) R[i] = red[0];
}

// ---------------------------------------------------------------------------
// K1/K3: H = relu(X @ W + b). Small (K<=64). Emits Hf (f32 row-major, for the
// +H residual term) and HT (bf16, transposed 64 x N, for WMMA B-fragments).
// ---------------------------------------------------------------------------
__global__ __launch_bounds__(256) void feat_gemm_kernel(
    const float* __restrict__ X, int ldx, int K,
    const float* __restrict__ W, const float* __restrict__ b,
    float* __restrict__ Hf, __bf16* __restrict__ HT)
{
    int idx = blockIdx.x * 256 + threadIdx.x;
    if (idx >= NNODES * 64) return;
    int row = idx >> 6;
    int c   = idx & 63;
    float s = b[c];
    for (int k = 0; k < K; ++k)
        s = fmaf(X[(size_t)row * ldx + k], W[k * 64 + c], s);
    s = fmaxf(s, 0.0f);
    Hf[idx] = s;
    HT[(size_t)c * NNODES + row] = (__bf16)s;
}

// ---------------------------------------------------------------------------
// Fragment set for one 16(rows) x 32(K) step against 4 column tiles.
// ---------------------------------------------------------------------------
struct Frags {
    bf16x8  alo, ahi;
    bf16x16 b[4];
};

__device__ __forceinline__ void load_frags(
    Frags& f, const __bf16* __restrict__ wrow, const __bf16* __restrict__ hbase,
    int k, int akb)
{
    f.alo = *(const bf16x8*)(wrow + k + akb);
    f.ahi = *(const bf16x8*)(wrow + k + akb + 8);
#pragma unroll
    for (int t = 0; t < 4; ++t)
        f.b[t] = *(const bf16x16*)(hbase + (size_t)(t * 16) * NNODES + k);
}

__device__ __forceinline__ void do_wmma(v8f* acc, const Frags& f)
{
    bf16x16 a;   // alo/ahi get allocated adjacently -> this combine is free
#pragma unroll
    for (int i = 0; i < 8; ++i) { a[i] = f.alo[i]; a[8 + i] = f.ahi[i]; }
#pragma unroll
    for (int t = 0; t < 4; ++t)
        acc[t] = __builtin_amdgcn_wmma_f32_16x16x32_bf16(
            false, a, false, f.b[t], (short)0, acc[t], false, false);
}

// ---------------------------------------------------------------------------
// K2: Z = relu( ((Wm @ H)/R + H) @ wt + bs )
// Block = 128 threads (4 waves). Each wave computes 16 rows x 64 cols of
// Y = Wm @ H with v_wmma_f32_16x16x32_bf16 (4 N-tiles, 256 K-steps).
// Triple-buffered rotation, unrolled by 3: each fragment set is reloaded one
// full WMMA-group AFTER its last read, so the VMEM writeback never targets
// registers an in-flight WMMA may still read — the copy/nop churn the
// 2-buffer version forced on the allocator has no reason to exist.
// ~152 VGPRs of live state; waves_per_eu(1) since this kernel runs ~1
// wave/SIMD chip-wide by design (512 waves total) and the RF is free.
// Epilogue: /R, +H, park Y in padded LDS, then f32 64x64 GEMM vs wt, relu.
// ---------------------------------------------------------------------------
__global__ __launch_bounds__(128)
__attribute__((amdgpu_waves_per_eu(1)))
void gcn_agg_wmma_kernel(
    const __bf16* __restrict__ Wm, const float* __restrict__ Rsum,
    const __bf16* __restrict__ HT, const float* __restrict__ Hf,
    const float* __restrict__ wt,  const float* __restrict__ bs,
    float* __restrict__ Z)
{
    __shared__ float Ylds[64][65];   // +1 pad: conflict-free column reads

    const int wave = threadIdx.x >> 5;
    const int lane = threadIdx.x & 31;
    const int blkrow = blockIdx.x * 64;
    const int row0 = blkrow + wave * 16;

    // WMMA fragment geometry (wave32, 16-bit; contiguous K order k = 16*(lane/16) + s, same for A and B)
    const int am  = lane & 15;          // A: row within tile
    const int akb = (lane >> 4) * 16;   // A: K sub-offset
    const int bn  = lane & 15;          // B: column within tile
    const int bks = (lane >> 4) * 16;   // B: K sub-offset

    v8f acc[4];
#pragma unroll
    for (int t = 0; t < 4; ++t)
#pragma unroll
        for (int i = 0; i < 8; ++i) acc[t][i] = 0.0f;

    const __bf16* wrow  = Wm + (size_t)(row0 + am) * NNODES;
    const __bf16* hbase = HT + (size_t)bn * NNODES + bks;

    // ---- triple-buffered K loop, 3 steps (96 K) per iteration ----
    Frags f0, f1, f2;
    load_frags(f0, wrow, hbase, 0, akb);
    load_frags(f1, wrow, hbase, 32, akb);
    int k0 = 0;
    for (; k0 + 160 <= NNODES; k0 += 96) {
        do_wmma(acc, f0);                            // step k0
        load_frags(f2, wrow, hbase, k0 + 64, akb);   // f2 idle for 1 group
        __builtin_prefetch(wrow + k0 + 4096, 0, 3);
        do_wmma(acc, f1);                            // step k0+32
        load_frags(f0, wrow, hbase, k0 + 96, akb);   // f0 idle for 1 group
        __builtin_prefetch(wrow + k0 + 4096 + 64, 0, 3);
        do_wmma(acc, f2);                            // step k0+64
        load_frags(f1, wrow, hbase, k0 + 128, akb);  // f1 idle for 1 group
    }
    // tail: k0 == 8064 here; f0 holds 8064, f1 holds 8096; 4 steps remain
    do_wmma(acc, f0);                                // 8064
    load_frags(f2, wrow, hbase, k0 + 64, akb);       // 8128
    do_wmma(acc, f1);                                // 8096
    load_frags(f0, wrow, hbase, k0 + 96, akb);       // 8160
    do_wmma(acc, f2);                                // 8128
    do_wmma(acc, f0);                                // 8160

    // Epilogue 1: Y = acc/R + Hf  -> LDS  (C layout: lanes0-15 M=r, 16-31 M=8+r)
#pragma unroll
    for (int r = 0; r < 8; ++r) {
        int lrl  = (lane < 16) ? r : (8 + r);
        int grow = row0 + lrl;
        float rinv = 1.0f / Rsum[grow];
#pragma unroll
        for (int t = 0; t < 4; ++t) {
            int col = t * 16 + (lane & 15);
            float y = acc[t][r] * rinv + Hf[(size_t)grow * 64 + col];
            Ylds[wave * 16 + lrl][col] = y;
        }
    }
    __syncthreads();

    // Epilogue 2: Z = relu(Y @ wt + bs). Each thread: 1 row, 32 cols.
    const int zr = threadIdx.x & 63;
    const int c0 = (threadIdx.x >> 6) * 32;
    float zacc[32];
    const float bsv = bs[0];
#pragma unroll
    for (int c = 0; c < 32; ++c) zacc[c] = bsv;
    for (int k = 0; k < 64; ++k) {
        float y = Ylds[zr][k];
        const float* wk = wt + k * 64 + c0;   // uniform across the wave
#pragma unroll
        for (int c = 0; c < 32; ++c) zacc[c] = fmaf(y, wk[c], zacc[c]);
    }
    float* zout = Z + (size_t)(blkrow + zr) * 64 + c0;
#pragma unroll
    for (int c = 0; c < 32; ++c) zout[c] = fmaxf(zacc[c], 0.0f);
}

// ---------------------------------------------------------------------------
// K4: s = V.sum(axis=0); out = sigmoid(s @ Wl + bl)
// ---------------------------------------------------------------------------
__global__ __launch_bounds__(64) void final_kernel(
    const float* __restrict__ V, const float* __restrict__ Wl,
    const float* __restrict__ bl, float* __restrict__ out)
{
    __shared__ float s[64];
    int c = threadIdx.x;
    float acc = 0.0f;
    for (int r = 0; r < NNODES; ++r) acc += V[(size_t)r * 64 + c];
    s[c] = acc * Wl[c];
    __syncthreads();
    if (c == 0) {
        float d = 0.0f;
        for (int k = 0; k < 64; ++k) d += s[k];
        d += bl[0];
        out[0] = 1.0f / (1.0f + __expf(-d));
    }
}

// ---------------------------------------------------------------------------
extern "C" void kernel_launch(void* const* d_in, const int* in_sizes, int n_in,
                              void* d_out, int out_size, void* d_ws, size_t ws_size,
                              hipStream_t stream)
{
    const float* x     = (const float*)d_in[0];
    const float* alpha = (const float*)d_in[1];
    const float* W1    = (const float*)d_in[2];
    const float* b1    = (const float*)d_in[3];
    const float* wt1   = (const float*)d_in[4];
    const float* bs1   = (const float*)d_in[5];
    const float* W2    = (const float*)d_in[6];
    const float* b2    = (const float*)d_in[7];
    const float* wt2   = (const float*)d_in[8];
    const float* bs2   = (const float*)d_in[9];
    const float* Wl    = (const float*)d_in[10];
    const float* bl    = (const float*)d_in[11];
    float* out = (float*)d_out;

    // Workspace carve (256B aligned); total ~138 MB
    char* p = (char*)d_ws;
    auto carve = [&](size_t bytes) {
        char* r = p;
        p += (bytes + 255) & ~(size_t)255;
        return (void*)r;
    };
    __bf16* Wm  = (__bf16*)carve((size_t)NNODES * NNODES * sizeof(__bf16));
    float*  R   = (float*)carve((size_t)NNODES * sizeof(float));
    float*  eta = (float*)carve((size_t)NNODES * sizeof(float));
    float*  phi = (float*)carve((size_t)NNODES * sizeof(float));
    float*  Hf1 = (float*)carve((size_t)NNODES * 64 * sizeof(float));
    __bf16* HT1 = (__bf16*)carve((size_t)NNODES * 64 * sizeof(__bf16));
    float*  Z1  = (float*)carve((size_t)NNODES * 64 * sizeof(float));
    float*  Hf2 = (float*)carve((size_t)NNODES * 64 * sizeof(float));
    __bf16* HT2 = (__bf16*)carve((size_t)NNODES * 64 * sizeof(__bf16));
    float*  V   = (float*)carve((size_t)NNODES * 64 * sizeof(float));

    // 1) eta/phi extraction
    extract_kernel<<<(NNODES + 255) / 256, 256, 0, stream>>>(x, eta, phi);
    // 2) edge/softmax-numerator matrix + rowsums (one block per row)
    edges_kernel<<<NNODES, 256, 0, stream>>>(eta, phi, alpha, Wm, R);
    // 3) layer 1 feature GEMM: H1 = relu(x @ W1 + b1)
    feat_gemm_kernel<<<(NNODES * 64 + 255) / 256, 256, 0, stream>>>(
        x, 7, 7, W1, b1, Hf1, HT1);
    // 4) layer 1 aggregation: Z1 = relu(((Wm@H1)/R + H1) @ wt1 + bs1)
    gcn_agg_wmma_kernel<<<NNODES / 64, 128, 0, stream>>>(
        Wm, R, HT1, Hf1, wt1, bs1, Z1);
    // 5) layer 2 feature GEMM: H2 = relu(Z1 @ W2 + b2)
    feat_gemm_kernel<<<(NNODES * 64 + 255) / 256, 256, 0, stream>>>(
        Z1, 64, 64, W2, b2, Hf2, HT2);
    // 6) layer 2 aggregation -> V
    gcn_agg_wmma_kernel<<<NNODES / 64, 128, 0, stream>>>(
        Wm, R, HT2, Hf2, wt2, bs2, V);
    // 7) column-sum + linear + sigmoid
    final_kernel<<<1, 64, 0, stream>>>(V, Wl, bl, out);
}
